// CNN_BiLSTM_65146063945742
// MI455X (gfx1250) — matrix operand order repaired and checked by reading, not compiled, not run on hardware
//
#include <hip/hip_runtime.h>
#include <hip/hip_bf16.h>

// ---------------- problem constants ----------------
#define B_    8
#define L_    4096
#define DIN_  128
#define LC_   4094            // L - 2 (VALID conv, k=3)
#define BL_   (B_ * LC_)      // 32752 rows, exact multiple of 16
#define DM_   64
#define DI_   128
#define DS_   16
#define DR_   4
#define H_    128
#define G4H_  512             // 4*H

typedef _Float16 f16;
typedef _Float16 h16v __attribute__((ext_vector_type(16)));
typedef float    f8v  __attribute__((ext_vector_type(8)));

// =====================================================================
// Generic WMMA GEMM:  out = epilogue( A(MxK,f16,row-major') @ W(NxK,f16)^T )
// A row base = m*aStride + (m/LC_)*aExtra  (aExtra!=0 only for the im2col
// conv view of x, where consecutive logical rows within a batch overlap).
// Weight buffers are zero-padded so every active wave's B loads are
// unconditionally in-bounds: the K-loop body is 4x global_load_b128 + wmma,
// no exec-mask churn. block = 128 threads (4 waves), one 16x16 N-tile per
// wave, grid = (M/16, ceil(N/64)).
// =====================================================================
__global__ __launch_bounds__(128)
void k_wmma_gemm(const f16* __restrict__ A, const f16* __restrict__ W,
                 const float* __restrict__ bias,
                 float* __restrict__ out32, f16* __restrict__ out16,
                 int N, int K, int aStride, int aExtra, int relu)
{
  const int lane  = threadIdx.x & 31;
  const int wave  = threadIdx.x >> 5;
  const int tileM = blockIdx.x * 16;
  const int nTile = blockIdx.y * 64 + wave * 16;
  if (nTile >= N) return;                 // uniform per wave

  const int r15 = lane & 15;
  const int hi  = lane >> 4;              // 0 or 1
  const int mRow = tileM + r15;
  const f16* ap = A + (size_t)mRow * aStride + (size_t)(mRow / LC_) * aExtra
                    + 8 * hi;
  const int nCol = nTile + r15;
  const f16* wp = W + (size_t)nCol * K + 8 * hi;    // padded weights: always OK

  f8v acc = {0.f,0.f,0.f,0.f,0.f,0.f,0.f,0.f};
  for (int k = 0; k < K; k += 32) {
    union { h16v v; uint4 q[2]; } a, b;
    // A 16x32 tile: lane<16 -> K k..k+7 & k+16..k+23 ; lane>=16 -> +8
    a.q[0] = *(const uint4*)(ap + k);
    a.q[1] = *(const uint4*)(ap + k + 16);
    // B 32x16 tile: lane<16 -> K k..k+7 & k+16..k+23 ; lane>=16 -> +8
    b.q[0] = *(const uint4*)(wp + k);
    b.q[1] = *(const uint4*)(wp + k + 16);
    acc = __builtin_amdgcn_wmma_f32_16x16x32_f16(false, a.v, false, b.v,
                                                 (short)0, acc, false, false);
  }
  if (nCol < N) {                         // only x_proj (N=36) ever masks here
    float bv = bias ? bias[nCol] : 0.f;
    #pragma unroll
    for (int r = 0; r < 8; ++r) {         // D: VGPR r, M = r + 8*hi, N = r15
      int m = tileM + r + 8 * hi;
      float v = acc[r] + bv;
      if (relu) v = v > 0.f ? v : 0.f;
      size_t o = (size_t)m * N + nCol;
      if (out32) out32[o] = v;
      if (out16) out16[o] = (f16)v;
    }
  }
}

// =====================================================================
// Persistent BiLSTM recurrence. grid = 2 (directions), 256 threads (8 waves).
// Whh (512x128 f16) + h (16x128 f16, rows 8..15 zero) + gates + c live in LDS
// (164 KB of the 320 KB WGP budget). Per step: z = gx[t] + h @ Whh^T via WMMA
// (each wave: 4 N-tiles x 4 K-tiles), then elementwise LSTM cell update.
// Next step's gx rows are prefetched (global_prefetch_b8) during compute.
// =====================================================================
#define WHH_BYTES (G4H_ * 136 * 2)        // 512 rows, stride 136 f16 = 139264
#define HB_OFF  WHH_BYTES                 // 16 x 136 f16 = 4352
#define ZS_OFF  (HB_OFF + 16*136*2)       // 8 x 512 f32  = 16384
#define CS_OFF  (ZS_OFF + 8*512*4)        // 8 x 128 f32  = 4096
#define REC_LDS (CS_OFF + 8*128*4)        // 164096 bytes total (< 320 KB/WGP)

__global__ __launch_bounds__(256)
void k_lstm_rec(const float* __restrict__ gxf, const float* __restrict__ gxb,
                const f16* __restrict__ whh16,    // (2, 512, 128) f16
                float* __restrict__ out32, f16* __restrict__ out16,
                int outStride)
{
  extern __shared__ char smem[];
  f16*   whh = (f16*)(smem);
  f16*   hb  = (f16*)(smem + HB_OFF);
  float* zs  = (float*)(smem + ZS_OFF);
  float* cst = (float*)(smem + CS_OFF);

  const int dir  = blockIdx.x;
  const int tid  = threadIdx.x;
  const int lane = tid & 31;
  const int wave = tid >> 5;
  const int r15  = lane & 15;
  const int hi   = lane >> 4;
  const float* gx = dir ? gxb : gxf;
  const uint4* wsrc = (const uint4*)(whh16 + (size_t)dir * G4H_ * H_);

  // stage Whh into LDS (row n, padded stride 272 B = 17 x 16 B, conflict-free)
  for (int idx = tid; idx < G4H_ * (H_ / 8); idx += 256) {
    int n = idx >> 4, q = idx & 15;
    *(uint4*)((char*)whh + n * 272 + q * 16) = wsrc[idx];
  }
  for (int i = tid; i < 16 * 136; i += 256) hb[i]  = (f16)0.f;
  for (int i = tid; i < 8 * 128;  i += 256) cst[i] = 0.f;
  __syncthreads();

  for (int t = 0; t < LC_; ++t) {
    const int ta = dir ? (LC_ - 1 - t) : t;

    // prefetch next step's gx rows (8 x 512 f32) while this step computes
    if (t + 1 < LC_ && tid < 128) {
      int tn = dir ? (LC_ - 2 - t) : (t + 1);
      int pm = tid >> 4, pc = (tid & 15) * 32;
      __builtin_prefetch(gx + ((size_t)pm * LC_ + tn) * G4H_ + pc, 0, 0);
    }

    // load the four 16x32 A (=h) tiles once per step
    union U { h16v v; uint4 q[2]; };
    U a[4];
    #pragma unroll
    for (int kt = 0; kt < 4; ++kt) {
      const f16* hp = hb + r15 * 136 + kt * 32;
      a[kt].q[0] = *(const uint4*)(hp +      8 * hi);
      a[kt].q[1] = *(const uint4*)(hp + 16 + 8 * hi);
    }

    #pragma unroll
    for (int j = 0; j < 4; ++j) {
      const int n = wave * 64 + j * 16 + r15;     // column in [0, 512)
      f8v acc;
      #pragma unroll
      for (int r = 0; r < 8; ++r) {               // C := gx[b, ta, n] (b = M row)
        int m = r + 8 * hi;
        acc[r] = (m < B_) ? gx[((size_t)m * LC_ + ta) * G4H_ + n] : 0.f;
      }
      #pragma unroll
      for (int kt = 0; kt < 4; ++kt) {
        U b;
        const char* wp = (char*)whh + n * 272 + (kt * 32 + 8 * hi) * 2;
        b.q[0] = *(const uint4*)(wp);
        b.q[1] = *(const uint4*)(wp + 32);
        acc = __builtin_amdgcn_wmma_f32_16x16x32_f16(false, a[kt].v, false, b.v,
                                                     (short)0, acc, false, false);
      }
      #pragma unroll
      for (int r = 0; r < 8; ++r) {
        int m = r + 8 * hi;
        if (m < B_) zs[m * G4H_ + n] = acc[r];
      }
    }
    __syncthreads();

    // LSTM cell: 8 x 128 cells, 4 per thread
    #pragma unroll
    for (int jj = 0; jj < 4; ++jj) {
      int idx = tid + 256 * jj;                   // 0..1023
      int m = idx >> 7, col = idx & 127;
      float iv = zs[m * 512 + col];
      float fv = zs[m * 512 + 128 + col];
      float gv = zs[m * 512 + 256 + col];
      float ov = zs[m * 512 + 384 + col];
      float c  = cst[m * 128 + col];
      float si = 1.f / (1.f + __expf(-iv));
      float sf = 1.f / (1.f + __expf(-fv));
      float so = 1.f / (1.f + __expf(-ov));
      c = sf * c + si * tanhf(gv);
      float h = so * tanhf(c);
      cst[m * 128 + col] = c;
      hb[m * 136 + col]  = (f16)h;
      size_t o = ((size_t)m * LC_ + ta) * outStride + dir * 128 + col;
      out32[o] = h;
      if (out16) out16[o] = (f16)h;
    }
    __syncthreads();
  }
}

// =====================================================================
// Elementwise / small kernels
// =====================================================================
__global__ void k_cvt16(const float* __restrict__ s, f16* __restrict__ d, int n) {
  int i = blockIdx.x * 256 + threadIdx.x;
  if (i < n) d[i] = (f16)s[i];
}
// convert with zero padding: d[i] = (i < nvalid) ? f16(s[i]) : 0, i < ntotal
__global__ void k_cvt16_pad(const float* __restrict__ s, f16* __restrict__ d,
                            int nvalid, int ntotal) {
  int i = blockIdx.x * 256 + threadIdx.x;
  if (i < ntotal) d[i] = (i < nvalid) ? (f16)s[i] : (f16)0.f;
}
__global__ void k_conv_reorder(const float* __restrict__ cw, f16* __restrict__ d) {
  int i = blockIdx.x * 256 + threadIdx.x;          // d[o][t*128+c] = cw[o][c][t]
  if (i >= 64 * 384) return;
  int o = i / 384, k = i % 384, t = k >> 7, c = k & 127;
  d[i] = (f16)cw[(o * 128 + c) * 3 + t];
}
__global__ void k_As(const float* __restrict__ alog, float* __restrict__ As) {
  int i = blockIdx.x * 256 + threadIdx.x;
  if (i < DI_ * DS_) As[i] = -__expf(alog[i]);
}
__global__ void k_bsum(const float* __restrict__ a, const float* __restrict__ b,
                       float* __restrict__ o, int n) {
  int i = blockIdx.x * 256 + threadIdx.x;
  if (i < n) o[i] = a[i] + b[i];
}
// depthwise causal conv (k=3, pad left 2) + SiLU; xm = xz[..., :128]
__global__ void k_dconv_silu(const float* __restrict__ xz,
                             const float* __restrict__ w, const float* __restrict__ b,
                             float* __restrict__ u32, f16* __restrict__ u16) {
  int idx = blockIdx.x * 256 + threadIdx.x;
  if (idx >= BL_ * DI_) return;
  int m = idx >> 7, d = idx & 127;
  int bb = m / LC_, l = m - bb * LC_;
  float s = b[d];
  #pragma unroll
  for (int t = 0; t < 3; ++t) {
    int lp = l - 2 + t;
    if (lp >= 0) s += xz[((size_t)(bb * LC_ + lp)) * 256 + d] * w[d * 3 + t];
  }
  float v = s / (1.f + __expf(-s));
  u32[idx] = v; u16[idx] = (f16)v;
}
// dt = softplus(dt_r @ dt_proj_w^T + b); dt_r = x_dbl[..., :4]
__global__ void k_dt(const float* __restrict__ xdbl, const float* __restrict__ w,
                     const float* __restrict__ bias, float* __restrict__ dt) {
  int idx = blockIdx.x * 256 + threadIdx.x;
  if (idx >= BL_ * DI_) return;
  int m = idx >> 7, d = idx & 127;
  const float* xr = xdbl + (size_t)m * 36;
  float s = bias[d];
  #pragma unroll
  for (int r = 0; r < DR_; ++r) s += xr[r] * w[d * DR_ + r];
  dt[idx] = (s > 20.f) ? s : log1pf(__expf(s));
}
// selective scan: grid (B, DI/32), 32 lanes, 16 states/lane in registers,
// next-step operand rows prefetched during the current step.
__global__ __launch_bounds__(32)
void k_scan(const float* __restrict__ dtb, const float* __restrict__ u32,
            const float* __restrict__ xz, const float* __restrict__ xdbl,
            const float* __restrict__ As, const float* __restrict__ Dp,
            f16* __restrict__ y16) {
  int b = blockIdx.x;
  int d = blockIdx.y * 32 + threadIdx.x;
  float Ar[DS_], st[DS_];
  #pragma unroll
  for (int n = 0; n < DS_; ++n) { Ar[n] = As[d * DS_ + n]; st[n] = 0.f; }
  const float Dpd = Dp[d];
  for (int l = 0; l < LC_; ++l) {
    size_t m = (size_t)b * LC_ + l;
    if (l + 1 < LC_) {
      __builtin_prefetch(dtb + (m + 1) * 128 + d, 0, 0);
      __builtin_prefetch(u32 + (m + 1) * 128 + d, 0, 0);
      __builtin_prefetch(xz + (m + 1) * 256 + 128 + d, 0, 0);
      __builtin_prefetch(xdbl + (m + 1) * 36, 0, 0);
    }
    float dt = dtb[m * 128 + d];
    float u  = u32[m * 128 + d];
    float z  = xz[m * 256 + 128 + d];
    const float* xr = xdbl + m * 36;
    float du = dt * u, y = 0.f;
    #pragma unroll
    for (int n = 0; n < DS_; ++n) {
      st[n] = __expf(dt * Ar[n]) * st[n] + du * xr[4 + n];   // Bm
      y += st[n] * xr[20 + n];                               // Cm
    }
    y += u * Dpd;
    float sz = z / (1.f + __expf(-z));                       // silu(z)
    y16[m * 128 + d] = (f16)(y * sz);
  }
}
__global__ void k_fc(const float* __restrict__ l1, const float* __restrict__ w,
                     const float* __restrict__ b, float* __restrict__ out) {
  int m = blockIdx.x * 256 + threadIdx.x;
  if (m >= BL_) return;
  const float* r = l1 + (size_t)m * 256;
  float s = b[0];
  for (int j = 0; j < 256; ++j) s += r[j] * w[j];
  out[m] = 1.f / (1.f + __expf(-s));
}

// =====================================================================
extern "C" void kernel_launch(void* const* d_in, const int* in_sizes, int n_in,
                              void* d_out, int out_size, void* d_ws, size_t ws_size,
                              hipStream_t stream) {
  const float* x        = (const float*)d_in[0];
  const float* conv_w   = (const float*)d_in[1];
  const float* conv_b   = (const float*)d_in[2];
  const float* in_proj  = (const float*)d_in[3];
  const float* dconv_w  = (const float*)d_in[4];
  const float* dconv_b  = (const float*)d_in[5];
  const float* x_proj   = (const float*)d_in[6];
  const float* dt_w     = (const float*)d_in[7];
  const float* dt_b     = (const float*)d_in[8];
  const float* A_log    = (const float*)d_in[9];
  const float* Dp       = (const float*)d_in[10];
  const float* out_proj = (const float*)d_in[11];
  const float* wih0     = (const float*)d_in[12];
  const float* whh0     = (const float*)d_in[13];
  const float* bih0     = (const float*)d_in[14];
  const float* bhh0     = (const float*)d_in[15];
  const float* wih1     = (const float*)d_in[16];
  const float* whh1     = (const float*)d_in[17];
  const float* bih1     = (const float*)d_in[18];
  const float* bhh1     = (const float*)d_in[19];
  const float* fc_w     = (const float*)d_in[20];
  const float* fc_b     = (const float*)d_in[21];
  float* outp = (float*)d_out;

  char* p = (char*)d_ws;
  auto carve = [&](size_t bytes) -> char* {
    char* r = p; p += (bytes + 255) & ~(size_t)255; return r;
  };
  f16*   x16    = (f16*)  carve((size_t)B_*L_*DIN_*2);     // 8.4 MB
  f16*   cw16   = (f16*)  carve(64*384*2);
  f16*   ip16   = (f16*)  carve(256*64*2);
  f16*   xp16   = (f16*)  carve(64*128*2);                 // padded 36 -> 64 rows
  f16*   op16   = (f16*)  carve(64*128*2);
  f16*   wih0_16= (f16*)  carve(2*512*64*2);
  f16*   whh0_16= (f16*)  carve(2*512*128*2);
  f16*   wih1_16= (f16*)  carve(2*512*256*2);
  f16*   whh1_16= (f16*)  carve(2*512*128*2);
  float* As     = (float*)carve(DI_*DS_*4);
  float* bsum0  = (float*)carve(1024*4);
  float* bsum1  = (float*)carve(1024*4);
  f16*   xc16   = (f16*)  carve((size_t)BL_*64*2);
  float* xz     = (float*)carve((size_t)BL_*256*4);        // 33.5 MB (reused as l1)
  float* u32    = (float*)carve((size_t)BL_*128*4);
  f16*   u16    = (f16*)  carve((size_t)BL_*128*2);
  float* xdbl   = (float*)carve((size_t)BL_*36*4);
  float* dtb    = (float*)carve((size_t)BL_*128*4);
  f16*   y16    = (f16*)  carve((size_t)BL_*128*2);
  f16*   xmo16  = (f16*)  carve((size_t)BL_*64*2);
  float* gxf    = (float*)carve((size_t)BL_*512*4);        // 67 MB (reused layer1)
  float* gxb    = (float*)carve((size_t)BL_*512*4);        // 67 MB
  float* l0     = (float*)carve((size_t)BL_*256*4);
  f16*   l0_16  = (f16*)  carve((size_t)BL_*256*2);
  float* l1     = xz;                                      // alias: xz dead by then
  (void)ws_size; (void)in_sizes; (void)n_in; (void)out_size;

  auto g1 = [](int n) { return dim3((n + 255) / 256); };

  // ---- weight conversions ----
  k_cvt16<<<g1(B_*L_*DIN_), 256, 0, stream>>>(x, x16, B_*L_*DIN_);
  k_conv_reorder<<<g1(64*384), 256, 0, stream>>>(conv_w, cw16);
  k_cvt16<<<g1(256*64), 256, 0, stream>>>(in_proj, ip16, 256*64);
  k_cvt16_pad<<<g1(64*128), 256, 0, stream>>>(x_proj, xp16, 36*128, 64*128);
  k_cvt16<<<g1(64*128), 256, 0, stream>>>(out_proj, op16, 64*128);
  k_cvt16<<<g1(2*512*64), 256, 0, stream>>>(wih0, wih0_16, 2*512*64);
  k_cvt16<<<g1(2*512*128), 256, 0, stream>>>(whh0, whh0_16, 2*512*128);
  k_cvt16<<<g1(2*512*256), 256, 0, stream>>>(wih1, wih1_16, 2*512*256);
  k_cvt16<<<g1(2*512*128), 256, 0, stream>>>(whh1, whh1_16, 2*512*128);
  k_As<<<g1(DI_*DS_), 256, 0, stream>>>(A_log, As);
  k_bsum<<<g1(1024), 256, 0, stream>>>(bih0, bhh0, bsum0, 1024);
  k_bsum<<<g1(1024), 256, 0, stream>>>(bih1, bhh1, bsum1, 1024);

  const int MT = BL_ / 16;   // 2047
  // ---- conv (im2col GEMM) + bias + relu -> xc16 ----
  k_wmma_gemm<<<dim3(MT, 1), 128, 0, stream>>>(x16, cw16, conv_b,
      nullptr, xc16, 64, 384, /*aStride*/128, /*aExtra*/(L_-LC_)*DIN_, 1);
  // ---- in_proj -> xz (f32, xm|z) ----
  k_wmma_gemm<<<dim3(MT, 4), 128, 0, stream>>>(xc16, ip16, nullptr,
      xz, nullptr, 256, 64, 64, 0, 0);
  // ---- depthwise conv + silu -> u ----
  k_dconv_silu<<<g1(BL_*DI_), 256, 0, stream>>>(xz, dconv_w, dconv_b, u32, u16);
  // ---- x_proj -> x_dbl (N=36, weights padded to 64 rows) ----
  k_wmma_gemm<<<dim3(MT, 1), 128, 0, stream>>>(u16, xp16, nullptr,
      xdbl, nullptr, 36, 128, 128, 0, 0);
  // ---- dt = softplus(...) ----
  k_dt<<<g1(BL_*DI_), 256, 0, stream>>>(xdbl, dt_w, dt_b, dtb);
  // ---- selective scan + Dp skip + silu(z) gate -> y16 ----
  k_scan<<<dim3(B_, DI_/32), 32, 0, stream>>>(dtb, u32, xz, xdbl, As, Dp, y16);
  // ---- out_proj -> xm_out (f16) ----
  k_wmma_gemm<<<dim3(MT, 1), 128, 0, stream>>>(y16, op16, nullptr,
      nullptr, xmo16, 64, 128, 128, 0, 0);
  // ---- LSTM layer 0: input projections then recurrence ----
  k_wmma_gemm<<<dim3(MT, 8), 128, 0, stream>>>(xmo16, wih0_16, bsum0,
      gxf, nullptr, 512, 64, 64, 0, 0);
  k_wmma_gemm<<<dim3(MT, 8), 128, 0, stream>>>(xmo16, wih0_16 + 512*64, bsum0 + 512,
      gxb, nullptr, 512, 64, 64, 0, 0);
  k_lstm_rec<<<dim3(2), 256, REC_LDS, stream>>>(gxf, gxb, whh0_16, l0, l0_16, 256);
  // ---- LSTM layer 1 ----
  k_wmma_gemm<<<dim3(MT, 8), 128, 0, stream>>>(l0_16, wih1_16, bsum1,
      gxf, nullptr, 512, 256, 256, 0, 0);
  k_wmma_gemm<<<dim3(MT, 8), 128, 0, stream>>>(l0_16, wih1_16 + 512*256, bsum1 + 512,
      gxb, nullptr, 512, 256, 256, 0, 0);
  k_lstm_rec<<<dim3(2), 256, REC_LDS, stream>>>(gxf, gxb, whh1_16, l1, nullptr, 256);
  // ---- fc + sigmoid -> out ----
  k_fc<<<g1(BL_), 256, 0, stream>>>(l1, fc_w, fc_b, outp);
}
